// IsotropicPolarizabilityDecoder_27625229648412
// MI455X (gfx1250) — matrix operand order repaired and checked by reading, not compiled, not run on hardware
//
#include <hip/hip_runtime.h>
#include <hip/hip_bf16.h>

typedef __attribute__((ext_vector_type(16))) _Float16 v16h;
typedef __attribute__((ext_vector_type(8)))  _Float16 v8h;
typedef __attribute__((ext_vector_type(8)))  float    v8f;

#define F 128
#define H 64
#define SRX 136   // LDS row stride for x-hat (halfs); 272B = 16*17, keeps 16B alignment
#define SRH 132   // LDS row stride for hidden (floats)

// ---------------------------------------------------------------------------
// Kernel 0: zero the output accumulator (graph replays don't re-poison).
// ---------------------------------------------------------------------------
__global__ void ipd_zero_kernel(float* __restrict__ out, int n) {
    int i = blockIdx.x * blockDim.x + threadIdx.x;
    if (i < n) out[i] = 0.0f;
}

// ---------------------------------------------------------------------------
// Kernel 1: fold LayerNorm affines into the first Linear layers and build the
// combined transposed f16 weight WcatT[col][k] (cols 0..63 = alpha branch,
// 64..127 = gate branch) plus fused bias biasCat[col].
//   WcatT[c][k] = g[k] * W1[k][c]          (f16)
//   biasCat[c]  = b1[c] + sum_k b[k]*W1[k][c]
// grid = 128 blocks (one per col), block = 128 threads (one per k).
// ---------------------------------------------------------------------------
__global__ void ipd_prep_kernel(const float* __restrict__ ln_a_g, const float* __restrict__ ln_a_b,
                                const float* __restrict__ Wa1,    const float* __restrict__ ba1,
                                const float* __restrict__ ln_n_g, const float* __restrict__ ln_n_b,
                                const float* __restrict__ Wn1,    const float* __restrict__ bn1,
                                _Float16* __restrict__ Wt, float* __restrict__ biasCat) {
    int c = blockIdx.x;   // 0..127 output column
    int k = threadIdx.x;  // 0..127 input feature
    float w, g, b, bias0;
    if (c < H) { w = Wa1[k * H + c];       g = ln_a_g[k]; b = ln_a_b[k]; bias0 = ba1[c]; }
    else       { w = Wn1[k * H + (c - H)]; g = ln_n_g[k]; b = ln_n_b[k]; bias0 = bn1[c - H]; }
    Wt[c * F + k] = (_Float16)(g * w);
    __shared__ float red[F];
    red[k] = b * w;
    __syncthreads();
    #pragma unroll
    for (int s = 64; s > 0; s >>= 1) {
        if (k < s) red[k] += red[k + s];
        __syncthreads();
    }
    if (k == 0) biasCat[c] = bias0 + red[0];
}

// ---------------------------------------------------------------------------
// Helpers
// ---------------------------------------------------------------------------
__device__ __forceinline__ float red8(float x) {
    x += __shfl_xor(x, 4, 32);
    x += __shfl_xor(x, 2, 32);
    x += __shfl_xor(x, 1, 32);
    return x;
}

// Fast SiLU: x * sigmoid(x) with hardware v_exp_f32 + v_rcp_f32 (no precise
// division chain — the GEMM feeding this is f16 anyway).
__device__ __forceinline__ float fast_silu(float x) {
    return x * __builtin_amdgcn_rcpf(1.0f + __expf(-x));
}

// A-fragment (16x32 f16) from LDS row-major xhat; matches ISA 16-bit A layout:
// lanes 0-15: K[0..7] in VGPR0-3, K[16..23] in VGPR4-7;
// lanes 16-31: K[8..15], K[24..31].
__device__ __forceinline__ v16h load_a_frag(const _Float16* sX, int kslice, int lane) {
    int row = lane & 15;
    int khi = (lane >> 4) << 3;  // 0 or 8
    const _Float16* base = sX + row * SRX + kslice + khi;
    v8h lo = *(const v8h*)(base);
    v8h hi = *(const v8h*)(base + 16);
    v16h a;
    #pragma unroll
    for (int i = 0; i < 8; ++i) { a[i] = lo[i]; a[i + 8] = hi[i]; }
    return a;
}

// B-fragment (32x16 f16) from global WcatT[col][k] (row stride F):
// lane l -> column (l&15); lanes 0-15 hold K[0..7], K[16..23];
// lanes 16-31 hold K[8..15], K[24..31] (same order as A).
__device__ __forceinline__ v16h load_b_frag(const _Float16* Wt, int colBase, int kslice, int lane) {
    int col = colBase + (lane & 15);
    int k0  = kslice + ((lane >> 4) << 3);
    v8h lo = *(const v8h*)(Wt + col * F + k0);
    v8h hi = *(const v8h*)(Wt + col * F + k0 + 16);
    v16h b;
    #pragma unroll
    for (int i = 0; i < 8; ++i) { b[i] = lo[i]; b[i + 8] = hi[i]; }
    return b;
}

// ---------------------------------------------------------------------------
// Kernel 2: main fused decoder. 128 threads = 4 waves; each wave owns 2
// column-tiles (32 cols) of the fused 128-col first layer. Grid-stride over
// 16-node tiles; B fragments live in registers across tiles.
// ---------------------------------------------------------------------------
__global__ void __launch_bounds__(128)
ipd_main_kernel(const float* __restrict__ pos,
                const float* __restrict__ mass_center,
                const float* __restrict__ scaler,
                const float* __restrict__ vector,
                const int*   __restrict__ batch_index,
                const _Float16* __restrict__ Wt,
                const float* __restrict__ biasCat,
                const float* __restrict__ Wa2, const float* __restrict__ ba2,
                const float* __restrict__ Wn2, const float* __restrict__ bn2,
                const float* __restrict__ Wnu,
                float* __restrict__ out,
                int N, int numTiles) {
    __shared__ _Float16 sX[16 * SRX];
    __shared__ float    sH[16 * SRH];
    __shared__ float    sWa2[H], sWn2[H], sWnu[F], sBias[F];

    const int tid  = threadIdx.x;
    const int lane = tid & 31;
    const int wave = tid >> 5;

    if (tid < H) { sWa2[tid] = Wa2[tid]; sWn2[tid] = Wn2[tid]; }
    sWnu[tid]  = Wnu[tid];
    sBias[tid] = biasCat[tid];
    const float ba2v = ba2[0];
    const float bn2v = bn2[0];

    // Preload this wave's B fragments: 2 column tiles x 4 k-slices (64 VGPRs).
    const int colBase0 = wave * 32;
    const int colBase1 = wave * 32 + 16;
    v16h bfrag0[4], bfrag1[4];
    #pragma unroll
    for (int kk = 0; kk < 4; ++kk) {
        bfrag0[kk] = load_b_frag(Wt, colBase0, kk * 32, lane);
        bfrag1[kk] = load_b_frag(Wt, colBase1, kk * 32, lane);
    }
    __syncthreads();

    for (int tile = blockIdx.x; tile < numTiles; tile += gridDim.x) {
        const long nodeBase = (long)tile * 16;

        // ---- Stage 1: load scaler rows, LayerNorm (no affine), f16 -> LDS
        {
            const int g = tid >> 3;                    // row 0..15
            const int j = tid & 7;                     // 16-elem chunk
            long node = nodeBase + g;
            if (node >= N) node = N - 1;               // clamped safe read
            const float4* s4 = (const float4*)(scaler + node * F + j * 16);
            float4 q0 = s4[0], q1 = s4[1], q2 = s4[2], q3 = s4[3];
            float v[16] = {q0.x,q0.y,q0.z,q0.w, q1.x,q1.y,q1.z,q1.w,
                           q2.x,q2.y,q2.z,q2.w, q3.x,q3.y,q3.z,q3.w};
            float s = 0.f, ss = 0.f;
            #pragma unroll
            for (int i = 0; i < 16; ++i) { s += v[i]; ss += v[i] * v[i]; }
            s  = red8(s);
            ss = red8(ss);
            const float mean = s * (1.0f / F);
            const float var  = ss * (1.0f / F) - mean * mean;
            const float inv  = rsqrtf(var + 1e-5f);
            _Float16* dst = sX + g * SRX + j * 16;
            #pragma unroll
            for (int i = 0; i < 16; ++i) dst[i] = (_Float16)((v[i] - mean) * inv);
        }
        __syncthreads();

        // ---- Stage 2: fused first layer, one 16x128x128 GEMM via WMMA f16
        v8f c0 = {}, c1 = {};
        #pragma unroll
        for (int kk = 0; kk < 4; ++kk) {
            v16h a = load_a_frag(sX, kk * 32, lane);
            c0 = __builtin_amdgcn_wmma_f32_16x16x32_f16(false, a, false, bfrag0[kk],
                                                        (short)0, c0, false, false);
            c1 = __builtin_amdgcn_wmma_f32_16x16x32_f16(false, a, false, bfrag1[kk],
                                                        (short)0, c1, false, false);
        }
        // bias + SiLU, scatter to LDS hidden (C layout: VGPR r -> M=r / r+8)
        {
            const int col  = lane & 15;
            const int rOff = (lane >> 4) << 3;
            #pragma unroll
            for (int r = 0; r < 8; ++r) {
                float x0 = fast_silu(c0[r] + sBias[colBase0 + col]);
                float x1 = fast_silu(c1[r] + sBias[colBase1 + col]);
                sH[(r + rOff) * SRH + colBase0 + col] = x0;
                sH[(r + rOff) * SRH + colBase1 + col] = x1;
            }
        }
        __syncthreads();

        // ---- Stage 3: second-layer dots + vector projection + trace reduce
        {
            const int g = tid >> 3;   // node within tile
            const int j = tid & 7;    // lane within node-group
            const long node   = nodeBase + g;
            const long nclamp = (node < N) ? node : (long)(N - 1);

            float aacc = 0.f, nacc = 0.f;
            const float* hrow = sH + g * SRH;
            #pragma unroll
            for (int i = 0; i < 8; ++i) {
                aacc += hrow[j * 8 + i]     * sWa2[j * 8 + i];
                nacc += hrow[H + j * 8 + i] * sWn2[j * 8 + i];
            }
            float vd0 = 0.f, vd1 = 0.f, vd2 = 0.f;
            const float* vp = vector + nclamp * (3 * F) + j * 16;
            #pragma unroll
            for (int i = 0; i < 16; ++i) {
                const float w = sWnu[j * 16 + i];
                vd0 += vp[i]         * w;
                vd1 += vp[F + i]     * w;
                vd2 += vp[2 * F + i] * w;
            }
            aacc = red8(aacc); nacc = red8(nacc);
            vd0 = red8(vd0);   vd1 = red8(vd1);   vd2 = red8(vd2);

            if (j == 0 && node < N) {
                const int bi = batch_index[node];
                const float alpha_s = aacc + ba2v;
                const float gate    = nacc + bn2v;
                const float m0 = pos[node * 3 + 0] - mass_center[bi * 3 + 0];
                const float m1 = pos[node * 3 + 1] - mass_center[bi * 3 + 1];
                const float m2 = pos[node * 3 + 2] - mass_center[bi * 3 + 2];
                // trace(alpha)/3 = alpha_s + (2/3) * gate * (vproj . mcv)
                const float contrib =
                    alpha_s + (2.0f / 3.0f) * gate * (vd0 * m0 + vd1 * m1 + vd2 * m2);
                atomicAdd(&out[bi], contrib);
            }
        }
        __syncthreads();
    }
}

// ---------------------------------------------------------------------------
extern "C" void kernel_launch(void* const* d_in, const int* in_sizes, int n_in,
                              void* d_out, int out_size, void* d_ws, size_t ws_size,
                              hipStream_t stream) {
    const float* pos         = (const float*)d_in[0];
    const float* mass_center = (const float*)d_in[1];
    const float* scaler      = (const float*)d_in[2];
    const float* vector      = (const float*)d_in[3];
    const int*   batch_index = (const int*)  d_in[4];
    const float* ln_a_g      = (const float*)d_in[5];
    const float* ln_a_b      = (const float*)d_in[6];
    const float* Wa1         = (const float*)d_in[7];
    const float* ba1         = (const float*)d_in[8];
    const float* Wa2         = (const float*)d_in[9];
    const float* ba2         = (const float*)d_in[10];
    const float* ln_n_g      = (const float*)d_in[11];
    const float* ln_n_b      = (const float*)d_in[12];
    const float* Wn1         = (const float*)d_in[13];
    const float* bn1         = (const float*)d_in[14];
    const float* Wn2         = (const float*)d_in[15];
    const float* bn2         = (const float*)d_in[16];
    const float* Wnu         = (const float*)d_in[17];
    float* out = (float*)d_out;

    const int N = in_sizes[0] / 3;
    const int B = in_sizes[1] / 3;

    _Float16* Wt    = (_Float16*)d_ws;
    float* biasCat  = (float*)((char*)d_ws + (size_t)F * F * sizeof(_Float16));

    ipd_zero_kernel<<<(B + 255) / 256, 256, 0, stream>>>(out, B);
    ipd_prep_kernel<<<F, F, 0, stream>>>(ln_a_g, ln_a_b, Wa1, ba1,
                                         ln_n_g, ln_n_b, Wn1, bn1,
                                         Wt, biasCat);
    const int numTiles = (N + 15) / 16;
    const int blocks   = numTiles < 4096 ? numTiles : 4096;
    ipd_main_kernel<<<blocks, 128, 0, stream>>>(pos, mass_center, scaler, vector,
                                                batch_index, Wt, biasCat,
                                                Wa2, ba2, Wn2, bn2, Wnu,
                                                out, N, numTiles);
}
